// MURAT_7894149890113
// MI455X (gfx1250) — matrix operand order repaired and checked by reading, not compiled, not run on hardware
//
#include <hip/hip_runtime.h>
#include <hip/hip_bf16.h>

typedef __attribute__((ext_vector_type(16))) __bf16 v16bf;
typedef __attribute__((ext_vector_type(8)))  float  v8f;

#define BM  128   // block M tile
#define BNT 128   // block N tile
#define BK  32    // K step
#define SA  40    // LDS row stride (elems): 32 + 8 pad (80B, 16B aligned)
#define SB  40

#define BATCH 16384
#define HID   1024
#define K0PAD 96
#define CHUNKS 64
#define EPS_BN 1e-5f

#if __has_builtin(__builtin_amdgcn_global_load_async_to_lds_b128) && \
    __has_builtin(__builtin_amdgcn_s_wait_asynccnt)
#define HAS_ASYNC_LDS 1
#else
#define HAS_ASYNC_LDS 0
#endif

#if HAS_ASYNC_LDS
// Builtin signature (from compiler diagnostic): pointee is a GCC-vector int4,
// global operand in AS1 ("__device__"), LDS operand in AS3 ("__shared__").
typedef int v4i_vs __attribute__((vector_size(16)));
typedef __attribute__((address_space(1))) v4i_vs g_v4i;
typedef __attribute__((address_space(3))) v4i_vs l_v4i;
#endif

// 16-byte global -> LDS copy; async (ASYNCcnt) when the gfx1250 builtin exists.
__device__ __forceinline__ void copy16_g2l(const __bf16* g, __bf16* l) {
#if HAS_ASYNC_LDS
    __builtin_amdgcn_global_load_async_to_lds_b128(
        (g_v4i*)(unsigned long long)g,
        (l_v4i*)(unsigned)(unsigned long long)l, 0, 0);
#else
    *(uint4*)l = *(const uint4*)g;
#endif
}

__device__ __forceinline__ void wait_g2l() {
#if HAS_ASYNC_LDS
    __builtin_amdgcn_s_wait_asynccnt(0);
#endif
}

// ---------------------------------------------------------------------------
// bf16 WMMA GEMM:  C[M,N] = A[M,K](bf16,row-major,lda) x Bt[N,K](bf16)^T + bias
// Block 128x128, 8 waves in 2x4; wave tile 64x32 = 4x2 v_wmma_f32_16x16x32_bf16.
// Double-buffered LDS fed by async global->LDS copies overlapped with WMMA.
// ---------------------------------------------------------------------------
__global__ __launch_bounds__(256)
void gemm_bf16_kernel(const __bf16* __restrict__ A, int lda,
                      const __bf16* __restrict__ Bt, int ldb,
                      const float* __restrict__ bias,
                      float* __restrict__ C, int N, int K) {
    __shared__ __bf16 As[2][BM * SA];
    __shared__ __bf16 Bs[2][BNT * SB];

    const int tid  = threadIdx.x;
    const int lane = tid & 31;
    const int wv   = tid >> 5;
    const int wm   = wv >> 2;        // 0..1 : 64-row slab
    const int wn   = wv & 3;         // 0..3 : 32-col slab
    const int m15  = lane & 15;
    const int hlf  = lane >> 4;

    const int rowBase = blockIdx.x * BM;
    const int colBase = blockIdx.y * BNT;

    // cooperative tile-load coords: 128 rows x 64B per tile, 32B per thread
    const int lr = tid >> 1;               // 0..127
    const int lc = (tid & 1) * 16;         // elem offset (16 elems = 32B via 2x b128)

    const __bf16* Ag = A  + (size_t)(rowBase + lr) * lda + lc;
    const __bf16* Bg = Bt + (size_t)(colBase + lr) * ldb + lc;
    __bf16* Al0 = &As[0][lr * SA + lc];
    __bf16* Bl0 = &Bs[0][lr * SB + lc];
    __bf16* Al1 = &As[1][lr * SA + lc];
    __bf16* Bl1 = &Bs[1][lr * SB + lc];

    v8f acc[4][2] = {};

    // prologue: stage 0 into buffer 0
    copy16_g2l(Ag, Al0);      copy16_g2l(Ag + 8, Al0 + 8);
    copy16_g2l(Bg, Bl0);      copy16_g2l(Bg + 8, Bl0 + 8);

    const int nk = K / BK;
    for (int kt = 0; kt < nk; ++kt) {
        const int cur = kt & 1;
        wait_g2l();            // stage kt resident in LDS
        __syncthreads();       // all waves see it; prior reads of other buffer done

        if (kt + 1 < nk) {     // overlap stage kt+1 copies with stage kt compute
            int k1 = (kt + 1) * BK;
            __bf16* Al = cur ? Al0 : Al1;
            __bf16* Bl = cur ? Bl0 : Bl1;
            copy16_g2l(Ag + k1,     Al);
            copy16_g2l(Ag + k1 + 8, Al + 8);
            copy16_g2l(Bg + k1,     Bl);
            copy16_g2l(Bg + k1 + 8, Bl + 8);
        }

        // Fragment loads per ISA VGPR layouts (16-bit A 16x32 / B 32x16, wave32)
        union Frag { uint4 q[2]; v16bf v; } fa[4], fb[2];
        #pragma unroll
        for (int i = 0; i < 4; ++i) {
            const __bf16* p = &As[cur][(wm * 64 + i * 16 + m15) * SA + hlf * 8];
            fa[i].q[0] = *(const uint4*)(p);        // K = 8*hlf .. +7      -> VGPR0-3
            fa[i].q[1] = *(const uint4*)(p + 16);   // K = 16+8*hlf .. +7   -> VGPR4-7
        }
        #pragma unroll
        for (int j = 0; j < 2; ++j) {
            const __bf16* p = &Bs[cur][(wn * 32 + j * 16 + m15) * SB + hlf * 8];
            fb[j].q[0] = *(const uint4*)(p);        // K = 8*hlf .. +7
            fb[j].q[1] = *(const uint4*)(p + 16);   // K = 16+8*hlf .. +7
        }

        #pragma unroll
        for (int i = 0; i < 4; ++i)
            #pragma unroll
            for (int j = 0; j < 2; ++j)
                acc[i][j] = __builtin_amdgcn_wmma_f32_16x16x32_bf16(
                    false, fa[i].v, false, fb[j].v, (short)0, acc[i][j], false, false);
    }

    // epilogue: C/D layout VGPR v -> M = v + 8*hlf, N = m15; bias fused
    #pragma unroll
    for (int i = 0; i < 4; ++i) {
        int row0 = rowBase + wm * 64 + i * 16 + hlf * 8;
        #pragma unroll
        for (int j = 0; j < 2; ++j) {
            int col = colBase + wn * 32 + j * 16 + m15;
            float bv = bias[col];
            #pragma unroll
            for (int v = 0; v < 8; ++v)
                C[(size_t)(row0 + v) * N + col] = acc[i][j][v] + bv;
        }
    }
}

// ---------------------------------------------------------------------------
// Weight convert: Wt[n*Kpad + k] = bf16(W[k*N + n]), zero-pad k >= Kreal
// ---------------------------------------------------------------------------
__global__ void wconv_kernel(const float* __restrict__ W, __bf16* __restrict__ Wt,
                             int Kpad, int Kreal, int N) {
    int idx = blockIdx.x * 256 + threadIdx.x;
    if (idx >= Kpad * N) return;
    int n = idx / Kpad, k = idx % Kpad;
    Wt[idx] = (__bf16)((k < Kreal) ? W[(size_t)k * N + n] : 0.0f);
}

// ---------------------------------------------------------------------------
// Build x0 = concat(link[40], spatial_W[gather][20], temporal_W[gather][20], pad16)
// ---------------------------------------------------------------------------
__global__ void build_x0_kernel(const float* __restrict__ link,
                                const float* __restrict__ sW,
                                const float* __restrict__ tW,
                                const int* __restrict__ sidx,
                                const int* __restrict__ tidx,
                                __bf16* __restrict__ x0) {
    int idx = blockIdx.x * 256 + threadIdx.x;
    if (idx >= BATCH * K0PAD) return;
    int r = idx / K0PAD, c = idx % K0PAD;
    float v;
    if (c < 40)       v = link[(size_t)r * 40 + c];
    else if (c < 60)  { int si = sidx[2*r] * 30 + sidx[2*r + 1]; v = sW[(size_t)si * 20 + (c - 40)]; }
    else if (c < 80)  { int ti = tidx[2*r] * 7  + tidx[2*r + 1]; v = tW[(size_t)ti * 20 + (c - 60)]; }
    else              v = 0.0f;
    x0[idx] = (__bf16)v;
}

__global__ void conv_bf16_kernel(const float* __restrict__ x, __bf16* __restrict__ y, int total) {
    int stride = gridDim.x * blockDim.x;
    for (int idx = blockIdx.x * blockDim.x + threadIdx.x; idx < total; idx += stride)
        y[idx] = (__bf16)x[idx];
}

// ---------------------------------------------------------------------------
// BN column stats, deterministic two-stage.
// ---------------------------------------------------------------------------
__global__ void colstats_kernel(const float* __restrict__ h,
                                float* __restrict__ ps, float* __restrict__ ps2) {
    int c  = blockIdx.x * 256 + threadIdx.x;
    int r0 = blockIdx.y * (BATCH / CHUNKS);
    float s = 0.f, s2 = 0.f;
    for (int r = r0; r < r0 + BATCH / CHUNKS; ++r) {
        float v = h[(size_t)r * HID + c];
        s += v; s2 += v * v;
    }
    ps [blockIdx.y * HID + c] = s;
    ps2[blockIdx.y * HID + c] = s2;
}

__global__ void bn_finalize_kernel(const float* __restrict__ ps, const float* __restrict__ ps2,
                                   const float* __restrict__ g, const float* __restrict__ be,
                                   float* __restrict__ scale, float* __restrict__ shift) {
    int c = blockIdx.x * 256 + threadIdx.x;
    float s = 0.f, s2 = 0.f;
    for (int k = 0; k < CHUNKS; ++k) { s += ps[k * HID + c]; s2 += ps2[k * HID + c]; }
    const float invM = 1.0f / (float)BATCH;
    float mean = s * invM;
    float var  = s2 * invM - mean * mean;            // biased variance (matches jnp.var)
    float sc   = g[c] * rsqrtf(var + EPS_BN);
    scale[c] = sc;
    shift[c] = be[c] - mean * sc;
}

__global__ void bnrelu_kernel(const float* __restrict__ h,
                              const float* __restrict__ scale, const float* __restrict__ shift,
                              __bf16* __restrict__ ybf, int total) {
    int stride = gridDim.x * blockDim.x;
    for (int idx = blockIdx.x * blockDim.x + threadIdx.x; idx < total; idx += stride) {
        int c = idx & (HID - 1);
        float y = fmaxf(scale[c] * h[idx] + shift[c], 0.0f);
        ybf[idx] = (__bf16)y;
    }
}

__global__ void bnrelu_add_kernel(const float* __restrict__ h,
                                  const float* __restrict__ scale, const float* __restrict__ shift,
                                  float* __restrict__ xf, __bf16* __restrict__ xbf, int total) {
    int stride = gridDim.x * blockDim.x;
    for (int idx = blockIdx.x * blockDim.x + threadIdx.x; idx < total; idx += stride) {
        int c = idx & (HID - 1);
        float y  = fmaxf(scale[c] * h[idx] + shift[c], 0.0f);
        float xn = xf[idx] + y;
        xf[idx]  = xn;
        xbf[idx] = (__bf16)xn;
    }
}

// ---------------------------------------------------------------------------
// lin1: out[r] = relu(dot(x[r,:], w) + b); one wave32 per row
// ---------------------------------------------------------------------------
__global__ __launch_bounds__(256)
void lin1_kernel(const float* __restrict__ x, const float* __restrict__ w,
                 const float* __restrict__ b, float* __restrict__ out) {
    int row  = blockIdx.x * 8 + (threadIdx.x >> 5);
    int lane = threadIdx.x & 31;
    const float* xr = x + (size_t)row * HID;
    float s = 0.f;
    for (int k = lane; k < HID; k += 32) s += xr[k] * w[k];
    for (int off = 16; off > 0; off >>= 1) s += __shfl_down(s, off, 32);
    if (lane == 0) out[row] = fmaxf(s + b[0], 0.0f);
}

// ---------------------------------------------------------------------------
// Laplacian sums (deterministic fixed-tree reductions, no atomics)
// ---------------------------------------------------------------------------
__global__ __launch_bounds__(256)
void lap_spatial_kernel(const float* __restrict__ A, const float* __restrict__ W,
                        float* __restrict__ partial) {
    __shared__ float Wi[20];
    __shared__ float red[256];
    int i = blockIdx.x;
    if (threadIdx.x < 20) Wi[threadIdx.x] = W[(size_t)i * 20 + threadIdx.x];
    __syncthreads();
    float s = 0.f;
    for (int j = threadIdx.x; j < 900; j += 256) {
        const float* wj = W + (size_t)j * 20;
        float sq = 0.f;
        #pragma unroll
        for (int e = 0; e < 20; ++e) { float d = Wi[e] - wj[e]; sq += d * d; }
        float dist = (sq > 0.f) ? sqrtf(sq) : 0.f;
        s += A[(size_t)i * 900 + j] * dist;
    }
    red[threadIdx.x] = s;
    __syncthreads();
    for (int off = 128; off > 0; off >>= 1) {
        if (threadIdx.x < off) red[threadIdx.x] += red[threadIdx.x + off];
        __syncthreads();
    }
    if (threadIdx.x == 0) partial[i] = red[0];
}

__global__ __launch_bounds__(256)
void lap_reduce_kernel(const float* __restrict__ partial, int n, float* __restrict__ out) {
    __shared__ float red[256];
    float s = 0.f;
    for (int k = threadIdx.x; k < n; k += 256) s += partial[k];
    red[threadIdx.x] = s;
    __syncthreads();
    for (int off = 128; off > 0; off >>= 1) {
        if (threadIdx.x < off) red[threadIdx.x] += red[threadIdx.x + off];
        __syncthreads();
    }
    if (threadIdx.x == 0) *out = red[0];
}

__global__ __launch_bounds__(256)
void lap_temporal_kernel(const float* __restrict__ A, const float* __restrict__ W,
                         float* __restrict__ out) {
    __shared__ float red[256];
    const int n = 31;
    float s = 0.f;
    for (int p = threadIdx.x; p < n * n; p += 256) {
        int i = p / n, j = p % n;
        float sq = 0.f;
        #pragma unroll
        for (int e = 0; e < 20; ++e) { float d = W[i * 20 + e] - W[j * 20 + e]; sq += d * d; }
        s += A[p] * ((sq > 0.f) ? sqrtf(sq) : 0.f);
    }
    red[threadIdx.x] = s;
    __syncthreads();
    for (int off = 128; off > 0; off >>= 1) {
        if (threadIdx.x < off) red[threadIdx.x] += red[threadIdx.x + off];
        __syncthreads();
    }
    if (threadIdx.x == 0) *out = red[0];
}

// ---------------------------------------------------------------------------
extern "C" void kernel_launch(void* const* d_in, const int* in_sizes, int n_in,
                              void* d_out, int out_size, void* d_ws, size_t ws_size,
                              hipStream_t stream) {
    // Input layout: setup_inputs() dict order, params dict flattened in insertion order.
    const float* link    = (const float*)d_in[0];
    const float* A_s     = (const float*)d_in[1];
    const float* A_t     = (const float*)d_in[2];
    const float* sW      = (const float*)d_in[3];
    const float* tW      = (const float*)d_in[4];
    const float* lin0_W  = (const float*)d_in[5];
    const float* lin0_b  = (const float*)d_in[6];
    // rb{i}: base 7 + (i-1)*8 : W1,b1,W2,b2,g1,be1,g2,be2
    const float* lin1_W  = (const float*)d_in[39];
    const float* lin1_b  = (const float*)d_in[40];
    const int*   sidx    = (const int*)d_in[41];
    const int*   tidx    = (const int*)d_in[42];

    float* out = (float*)d_out;    // [0..16383] final, [16384] summ_s, [16385] summ_t

    char* wp = (char*)d_ws;
    auto carve = [&](size_t bytes) -> char* {
        char* p = wp;
        wp += (bytes + 255) & ~(size_t)255;
        return p;
    };
    __bf16* Wt_lin0 = (__bf16*)carve((size_t)HID * K0PAD * 2);
    __bf16* Wt_rb[8];
    for (int i = 0; i < 8; ++i) Wt_rb[i] = (__bf16*)carve((size_t)HID * HID * 2);
    __bf16* x0_bf  = (__bf16*)carve((size_t)BATCH * K0PAD * 2);
    __bf16* x_bf   = (__bf16*)carve((size_t)BATCH * HID * 2);
    __bf16* h1_bf  = (__bf16*)carve((size_t)BATCH * HID * 2);
    float*  xf32   = (float*)carve((size_t)BATCH * HID * 4);
    float*  hbuf   = (float*)carve((size_t)BATCH * HID * 4);
    float*  ps     = (float*)carve((size_t)CHUNKS * HID * 4);
    float*  ps2    = (float*)carve((size_t)CHUNKS * HID * 4);
    float*  scale  = (float*)carve(HID * 4);
    float*  shift  = (float*)carve(HID * 4);
    float*  lappar = (float*)carve(1024 * 4);

    const int total = BATCH * HID;
    const dim3 gemmGrid(BATCH / BM, HID / BNT);
    const int ELT_GRID = 4096;

    // --- weight preprocessing (bf16, transposed) ---
    wconv_kernel<<<(K0PAD * HID + 255) / 256, 256, 0, stream>>>(lin0_W, Wt_lin0, K0PAD, 80, HID);
    for (int i = 1; i <= 4; ++i) {
        int pb = 7 + (i - 1) * 8;
        wconv_kernel<<<(HID * HID + 255) / 256, 256, 0, stream>>>(
            (const float*)d_in[pb + 0], Wt_rb[(i - 1) * 2 + 0], HID, HID, HID);
        wconv_kernel<<<(HID * HID + 255) / 256, 256, 0, stream>>>(
            (const float*)d_in[pb + 2], Wt_rb[(i - 1) * 2 + 1], HID, HID, HID);
    }

    // --- build x0 and lin0 ---
    build_x0_kernel<<<(BATCH * K0PAD + 255) / 256, 256, 0, stream>>>(link, sW, tW, sidx, tidx, x0_bf);
    gemm_bf16_kernel<<<gemmGrid, 256, 0, stream>>>(x0_bf, K0PAD, Wt_lin0, K0PAD, lin0_b, xf32, HID, K0PAD);
    conv_bf16_kernel<<<ELT_GRID, 256, 0, stream>>>(xf32, x_bf, total);

    // --- residual blocks ---
    for (int i = 1; i <= 4; ++i) {
        int pb = 7 + (i - 1) * 8;
        const float* b1  = (const float*)d_in[pb + 1];
        const float* b2  = (const float*)d_in[pb + 3];
        const float* g1  = (const float*)d_in[pb + 4];
        const float* be1 = (const float*)d_in[pb + 5];
        const float* g2  = (const float*)d_in[pb + 6];
        const float* be2 = (const float*)d_in[pb + 7];

        gemm_bf16_kernel<<<gemmGrid, 256, 0, stream>>>(x_bf, HID, Wt_rb[(i-1)*2+0], HID, b1, hbuf, HID, HID);
        colstats_kernel<<<dim3(HID / 256, CHUNKS), 256, 0, stream>>>(hbuf, ps, ps2);
        bn_finalize_kernel<<<HID / 256, 256, 0, stream>>>(ps, ps2, g1, be1, scale, shift);
        bnrelu_kernel<<<ELT_GRID, 256, 0, stream>>>(hbuf, scale, shift, h1_bf, total);

        gemm_bf16_kernel<<<gemmGrid, 256, 0, stream>>>(h1_bf, HID, Wt_rb[(i-1)*2+1], HID, b2, hbuf, HID, HID);
        colstats_kernel<<<dim3(HID / 256, CHUNKS), 256, 0, stream>>>(hbuf, ps, ps2);
        bn_finalize_kernel<<<HID / 256, 256, 0, stream>>>(ps, ps2, g2, be2, scale, shift);
        bnrelu_add_kernel<<<ELT_GRID, 256, 0, stream>>>(hbuf, scale, shift, xf32, x_bf, total);
    }

    // --- lin1 + relu ---
    lin1_kernel<<<BATCH / 8, 256, 0, stream>>>(xf32, lin1_W, lin1_b, out);

    // --- laplacian sums ---
    lap_spatial_kernel<<<900, 256, 0, stream>>>(A_s, sW, lappar);
    lap_reduce_kernel<<<1, 256, 0, stream>>>(lappar, 900, out + BATCH);
    lap_temporal_kernel<<<1, 256, 0, stream>>>(A_t, tW, out + BATCH + 1);
}
